// DeepSetModule_747324309661
// MI455X (gfx1250) — matrix operand order repaired and checked by reading, not compiled, not run on hardware
//
#include <hip/hip_runtime.h>

// ---------------------------------------------------------------------------
// CDNA5 (gfx1250) DeepSet: fused bf16-WMMA MLP pipeline.
//   phi: [B*N,64] -> 256 -> 256 -> 256 (relu,relu,linear), masked pool over N
//   rho: [B,256] -> 256 -> 256 -> 128 (relu,relu,linear), zero invalid rows
// ---------------------------------------------------------------------------

typedef __attribute__((ext_vector_type(16))) __bf16 v16bf;
typedef __attribute__((ext_vector_type(8)))  float  v8f;

#define ACT_STRIDE 264   // bf16 elements per LDS activation row (256 + 8 pad)
                         // row pitch = 528 B (16B aligned, non-power-of-2 banks)

// f32 -> bf16 bits, round-to-nearest-even
static __device__ inline unsigned short f2bf(float f) {
    unsigned int u = __float_as_uint(f);
    unsigned int r = (u + 0x7FFFu + ((u >> 16) & 1u)) >> 16;
    return (unsigned short)r;
}
static __device__ inline unsigned int f2bf_pk(float lo, float hi) {
    return (unsigned int)f2bf(lo) | ((unsigned int)f2bf(hi) << 16);
}

// ---------------------------------------------------------------------------
// Pack a f32 row-major weight W[K][N] into bf16 WMMA B-fragment layout:
// fragment f = ntile*(K/32)+kstep holds a 32x16 tile; within a fragment,
// lane l stores 16 contiguous bf16 at offset f*512 + l*16, element e maps to
// W[kstep*32 + ((e&8)?16:0) + ((l&16)?8:0) + (e&7)][ntile*16 + (l&15)].
// ---------------------------------------------------------------------------
__global__ __launch_bounds__(256) void pack_weight_kernel(
    const float* __restrict__ src, unsigned short* __restrict__ dst, int K, int N)
{
    int idx = blockIdx.x * 256 + threadIdx.x;
    int total = K * N;
    if (idx >= total) return;
    int frag = idx >> 9;
    int r    = idx & 511;
    int lane = r >> 4;
    int e    = r & 15;
    int ksteps = K >> 5;
    int ntile  = frag / ksteps;
    int kstep  = frag - ntile * ksteps;
    int k = kstep * 32 + ((e & 8) ? 16 : 0) + ((lane & 16) ? 8 : 0) + (e & 7);
    int n = ntile * 16 + (lane & 15);
    dst[idx] = f2bf(src[k * N + n]);
}

// ---------------------------------------------------------------------------
// One MLP layer on a 16-row tile owned by this wave:
//   acc[t] (16x16 f32, t = ntile) = actIn[16 x 32*KSTEPS] * W + bias
// actIn: LDS bf16 rows of ACT_STRIDE; Wp: packed bf16 fragments in global.
// First k-step uses a shared zero C (no per-tile bias broadcast); bias is
// added in the epilogue where it dual-issues with downstream VALU.
// ---------------------------------------------------------------------------
template <int KSTEPS, int NTILES>
static __device__ inline void gemm_block(
    const unsigned short* __restrict__ actIn,
    const unsigned short* __restrict__ Wp,
    const float* __restrict__ bias,
    int lane, int m0, v8f* acc)
{
    const int kbaseLane = (lane & 16) ? 8 : 0;       // A-layout K offset per half-wave
    const int rowA      = m0 + (lane & 15);          // A-layout: M = lane % 16
    v8f czero = {};
#pragma unroll
    for (int ks = 0; ks < KSTEPS; ++ks) {
        if (ks + 1 < KSTEPS)                         // warm next k-step's fragments
            __builtin_prefetch(Wp + (size_t)(ks + 1) * 512, 0, 0);
        union { v16bf v; unsigned int d[8]; } A;
#pragma unroll
        for (int vv = 0; vv < 8; ++vv) {
            int k = ks * 32 + ((vv & 4) ? 16 : 0) + kbaseLane + (vv & 3) * 2;
            A.d[vv] = *(const unsigned int*)(actIn + rowA * ACT_STRIDE + k);
        }
#pragma unroll
        for (int t = 0; t < NTILES; ++t) {
            union { v16bf v; uint4 q[2]; } B;
            const uint4* wq = (const uint4*)(Wp + (size_t)(t * KSTEPS + ks) * 512 + lane * 16);
            B.q[0] = wq[0];
            B.q[1] = wq[1];
            acc[t] = __builtin_amdgcn_wmma_f32_16x16x32_bf16(
                false, A.v, false, B.v, (short)0,
                (ks == 0) ? czero : acc[t], false, false);
        }
    }
#pragma unroll
    for (int t = 0; t < NTILES; ++t) {
        float bv = bias[t * 16 + (lane & 15)];
#pragma unroll
        for (int i = 0; i < 8; ++i) acc[t][i] += bv;
    }
}

// Write acc tiles (relu'd) back to LDS activations as bf16, C/D layout:
// VGPR r: M = r + (lane>=16 ? 8 : 0), N = lane % 16.
template <int NTILES>
static __device__ inline void store_act_relu(
    unsigned short* __restrict__ actOut, const v8f* acc, int lane, int m0)
{
    const int mAdd = (lane & 16) ? 8 : 0;
    const int nLo  = lane & 15;
#pragma unroll
    for (int t = 0; t < NTILES; ++t) {
#pragma unroll
        for (int r = 0; r < 8; ++r) {
            float v = fmaxf(acc[t][r], 0.0f);
            actOut[(m0 + r + mAdd) * ACT_STRIDE + t * 16 + nLo] = f2bf(v);
        }
    }
}

// ---------------------------------------------------------------------------
// phi kernel: block = (batch b, row-tile of 128). 256 threads = 8 waves,
// wave w owns rows [w*16, w*16+16). Fused 3-layer MLP, then masked column
// sums reduced across waves in LDS; one atomicAdd per column per WG.
// ---------------------------------------------------------------------------
__global__ __launch_bounds__(256) void phi_kernel(
    const float* __restrict__ x,        // [256][1024][64]
    const int*   __restrict__ mask,     // [256][1024]
    const unsigned short* __restrict__ pw1p,
    const float* __restrict__ pb1,
    const unsigned short* __restrict__ pw2p,
    const float* __restrict__ pb2,
    const unsigned short* __restrict__ pw3p,
    const float* __restrict__ pb3,
    float* __restrict__ pooled,         // [256][256]
    int*   __restrict__ valid)          // [256]
{
    extern __shared__ __align__(16) char smem[];
    unsigned short* act = (unsigned short*)smem;    // [128][ACT_STRIDE] bf16

    const int b    = blockIdx.x;        // 0..255
    const int tile = blockIdx.y;        // 0..7
    const int tid  = threadIdx.x;
    const int lane = tid & 31;
    const int wave = tid >> 5;
    const int m0   = wave * 16;
    const size_t rowBase = (size_t)b * 1024 + (size_t)tile * 128;

    // Stage x tile -> LDS bf16 (each thread: one half-row of 32 f32, 4x b128)
    {
        int row     = tid >> 1;
        int colBase = (tid & 1) * 32;
        const float4* src = (const float4*)(x + (rowBase + row) * 64 + colBase);
        uint4* dstq = (uint4*)(act + row * ACT_STRIDE + colBase);
#pragma unroll
        for (int j = 0; j < 4; ++j) {
            float4 f0 = src[j * 2 + 0];
            float4 f1 = src[j * 2 + 1];
            uint4 p;
            p.x = f2bf_pk(f0.x, f0.y);
            p.y = f2bf_pk(f0.z, f0.w);
            p.z = f2bf_pk(f1.x, f1.y);
            p.w = f2bf_pk(f1.z, f1.w);
            dstq[j] = p;
        }
        if (tid < 128) {
            if (mask[rowBase + tid] != 0) atomicOr(valid + b, 1);
        }
    }
    __syncthreads();

    v8f acc[16];

    // Layer 1: 64 -> 256, relu
    gemm_block<2, 16>(act, pw1p, pb1, lane, m0, acc);
    __syncthreads();
    store_act_relu<16>(act, acc, lane, m0);
    __syncthreads();

    // Layer 2: 256 -> 256, relu
    gemm_block<8, 16>(act, pw2p, pb2, lane, m0, acc);
    __syncthreads();
    store_act_relu<16>(act, acc, lane, m0);
    __syncthreads();

    // Layer 3: 256 -> 256, linear; masked column-sum over this WG's 128 rows
    gemm_block<8, 16>(act, pw3p, pb3, lane, m0, acc);

    const int mAdd = (lane & 16) ? 8 : 0;
    float mv[8];
#pragma unroll
    for (int r = 0; r < 8; ++r)
        mv[r] = (mask[rowBase + m0 + r + mAdd] != 0) ? 1.0f : 0.0f;

    __syncthreads();                    // act reads done; reuse LDS for reduce
    float* red = (float*)smem;          // [8 waves][256 columns]
#pragma unroll
    for (int t = 0; t < 16; ++t) {
        float p = 0.0f;
#pragma unroll
        for (int r = 0; r < 8; ++r) p += acc[t][r] * mv[r];
        p += __shfl_xor(p, 16, 32);     // combine the two half-waves
        if (lane < 16) red[wave * 256 + t * 16 + lane] = p;
    }
    __syncthreads();

    float s = 0.0f;
#pragma unroll
    for (int w = 0; w < 8; ++w) s += red[w * 256 + tid];
    atomicAdd(pooled + b * 256 + tid, s);   // 256 atomics per WG (was 4096)
}

// ---------------------------------------------------------------------------
// rho kernel: 2 blocks x 128 rows. pooled(f32) -> bf16 LDS -> 3 WMMA layers,
// final 256->128 writes d_out with invalid batch rows forced to zero.
// ---------------------------------------------------------------------------
__global__ __launch_bounds__(256) void rho_kernel(
    const float* __restrict__ pooled,   // [256][256]
    const int*   __restrict__ valid,    // [256]
    const unsigned short* __restrict__ rw1p,
    const float* __restrict__ rb1,
    const unsigned short* __restrict__ rw2p,
    const float* __restrict__ rb2,
    const unsigned short* __restrict__ rw3p,
    const float* __restrict__ rb3,
    float* __restrict__ out)            // [256][128]
{
    extern __shared__ __align__(16) char smem[];
    unsigned short* act = (unsigned short*)smem;    // [128][ACT_STRIDE] bf16

    const int tile = blockIdx.x;        // 0..1
    const int tid  = threadIdx.x;
    const int lane = tid & 31;
    const int wave = tid >> 5;
    const int m0   = wave * 16;
    const int rowBase = tile * 128;

    // Stage pooled rows -> LDS bf16 (each thread: half-row of 128 f32, 16x b128)
    {
        int row     = tid >> 1;
        int colBase = (tid & 1) * 128;
        const float4* src = (const float4*)(pooled + (size_t)(rowBase + row) * 256 + colBase);
        uint4* dstq = (uint4*)(act + row * ACT_STRIDE + colBase);
#pragma unroll
        for (int j = 0; j < 16; ++j) {
            float4 f0 = src[j * 2 + 0];
            float4 f1 = src[j * 2 + 1];
            uint4 p;
            p.x = f2bf_pk(f0.x, f0.y);
            p.y = f2bf_pk(f0.z, f0.w);
            p.z = f2bf_pk(f1.x, f1.y);
            p.w = f2bf_pk(f1.z, f1.w);
            dstq[j] = p;
        }
    }
    __syncthreads();

    v8f acc[16];

    // Layer 1: 256 -> 256, relu
    gemm_block<8, 16>(act, rw1p, rb1, lane, m0, acc);
    __syncthreads();
    store_act_relu<16>(act, acc, lane, m0);
    __syncthreads();

    // Layer 2: 256 -> 256, relu
    gemm_block<8, 16>(act, rw2p, rb2, lane, m0, acc);
    __syncthreads();
    store_act_relu<16>(act, acc, lane, m0);
    __syncthreads();

    // Layer 3: 256 -> 128, linear; write output with invalid-row zeroing
    v8f acc2[8];
    gemm_block<8, 8>(act, rw3p, rb3, lane, m0, acc2);

    const int mAdd = (lane & 16) ? 8 : 0;
    const int nLo  = lane & 15;
#pragma unroll
    for (int t = 0; t < 8; ++t) {
#pragma unroll
        for (int r = 0; r < 8; ++r) {
            int row = rowBase + m0 + r + mAdd;
            float v = (valid[row] != 0) ? acc2[t][r] : 0.0f;
            out[(size_t)row * 128 + t * 16 + nLo] = v;
        }
    }
}

// ---------------------------------------------------------------------------
// Host launcher.
// Inputs: 0:x 1:mask 2:pw1 3:pb1 4:pw2 5:pb2 6:pw3 7:pb3
//         8:rw1 9:rb1 10:rw2 11:rb2 12:rw3 13:rb3
// Workspace layout (bytes):
//   [0)        pw1p bf16 16384  ->  32768
//   [32768)    pw2p bf16 65536  -> 131072
//   [163840)   pw3p bf16 65536  -> 131072
//   [294912)   rw1p bf16 65536  -> 131072
//   [425984)   rw2p bf16 65536  -> 131072
//   [557056)   rw3p bf16 32768  ->  65536
//   [622592)   pooled f32 256*256 -> 262144
//   [884736)   valid  i32 256     ->   1024
// ---------------------------------------------------------------------------
extern "C" void kernel_launch(void* const* d_in, const int* in_sizes, int n_in,
                              void* d_out, int out_size, void* d_ws, size_t ws_size,
                              hipStream_t stream)
{
    (void)in_sizes; (void)n_in; (void)out_size; (void)ws_size;

    const float* x    = (const float*)d_in[0];
    const int*   mask = (const int*)d_in[1];
    const float* pw1  = (const float*)d_in[2];
    const float* pb1  = (const float*)d_in[3];
    const float* pw2  = (const float*)d_in[4];
    const float* pb2  = (const float*)d_in[5];
    const float* pw3  = (const float*)d_in[6];
    const float* pb3  = (const float*)d_in[7];
    const float* rw1  = (const float*)d_in[8];
    const float* rb1  = (const float*)d_in[9];
    const float* rw2  = (const float*)d_in[10];
    const float* rb2  = (const float*)d_in[11];
    const float* rw3  = (const float*)d_in[12];
    const float* rb3  = (const float*)d_in[13];

    char* ws = (char*)d_ws;
    unsigned short* pw1p = (unsigned short*)(ws + 0);
    unsigned short* pw2p = (unsigned short*)(ws + 32768);
    unsigned short* pw3p = (unsigned short*)(ws + 163840);
    unsigned short* rw1p = (unsigned short*)(ws + 294912);
    unsigned short* rw2p = (unsigned short*)(ws + 425984);
    unsigned short* rw3p = (unsigned short*)(ws + 557056);
    float* pooled        = (float*)(ws + 622592);
    int*   valid         = (int*)(ws + 884736);

    // Pack all weights into WMMA B-fragment bf16 layout (L2-resident, ~608KB).
    pack_weight_kernel<<<(64 * 256 + 255) / 256, 256, 0, stream>>>(pw1, pw1p, 64, 256);
    pack_weight_kernel<<<(256 * 256 + 255) / 256, 256, 0, stream>>>(pw2, pw2p, 256, 256);
    pack_weight_kernel<<<(256 * 256 + 255) / 256, 256, 0, stream>>>(pw3, pw3p, 256, 256);
    pack_weight_kernel<<<(256 * 256 + 255) / 256, 256, 0, stream>>>(rw1, rw1p, 256, 256);
    pack_weight_kernel<<<(256 * 256 + 255) / 256, 256, 0, stream>>>(rw2, rw2p, 256, 256);
    pack_weight_kernel<<<(256 * 128 + 255) / 256, 256, 0, stream>>>(rw3, rw3p, 256, 128);

    // Zero the atomic accumulation buffers (pooled + valid).
    hipMemsetAsync(ws + 622592, 0, 262144 + 1024, stream);

    const size_t smemBytes = (size_t)128 * ACT_STRIDE * sizeof(unsigned short); // 67584

    dim3 phiGrid(256, 8);
    phi_kernel<<<phiGrid, 256, smemBytes, stream>>>(
        x, mask, pw1p, pb1, pw2p, pb2, pw3p, pb3, pooled, valid);

    rho_kernel<<<2, 256, smemBytes, stream>>>(
        pooled, valid, rw1p, rb1, rw2p, rb2, rw3p, rb3, (float*)d_out);
}
